// MultiHeadAttention_47047071760672
// MI455X (gfx1250) — matrix operand order repaired and checked by reading, not compiled, not run on hardware
//
#include <hip/hip_runtime.h>

// ---------------------------------------------------------------------------
// MHA forward for MI455X (gfx1250): all GEMMs on v_wmma_f32_16x16x32_bf16,
// flash-attention inner loop (scores never hit HBM), wave32 layouts per
// CDNA5 ISA 7.12.2.  Staging uses GLOBAL_LOAD_ASYNC_TO_LDS (ASYNCcnt) and
// the Tensor Data Mover (TENSORcnt) where the copy is conversion-free.
// ---------------------------------------------------------------------------

typedef __attribute__((ext_vector_type(16))) __bf16 v16bf;
typedef __attribute__((ext_vector_type(8)))  float  v8f;
typedef __attribute__((ext_vector_type(4)))  unsigned int u32x4;
typedef __attribute__((ext_vector_type(8)))  int i32x8;
typedef __attribute__((ext_vector_type(4)))  int i32x4;

#define AS1 __attribute__((address_space(1)))
#define AS3 __attribute__((address_space(3)))

#if defined(__AMDGCN__) && __has_builtin(__builtin_amdgcn_global_load_async_to_lds_b128) && __has_builtin(__builtin_amdgcn_s_wait_asynccnt)
#define USE_ASYNC_LDS 1
#else
#define USE_ASYNC_LDS 0
#endif

#if defined(__AMDGCN__) && __has_builtin(__builtin_amdgcn_tensor_load_to_lds) && __has_builtin(__builtin_amdgcn_s_wait_tensorcnt)
#define USE_TDM 1
#else
#define USE_TDM 0
#endif

union Frag16 { unsigned int u[8]; v16bf v; };

__device__ __forceinline__ unsigned short bf16_of(float a) {
  unsigned int ua = __float_as_uint(a);
  return (unsigned short)((ua + 0x7FFFu + ((ua >> 16) & 1u)) >> 16);
}
__device__ __forceinline__ unsigned int pack_bf16(float a, float b) {
  return (unsigned int)bf16_of(a) | ((unsigned int)bf16_of(b) << 16);
}

#define MHA_S  2048
#define MHA_D  1024
#define MHA_H  16
#define MHA_HD 64

// ---------------------------------------------------------------------------
// Kernel 1: fused QKV projection.  out = x @ W{q,k,v}, stored bf16 in
// (B,H,S,HD) layout.  Block tile 128(M) x 64(N), 8 waves, K-step 32.
// (A/B need f32->bf16 conversion in flight, so staging stays in VGPRs.)
// ---------------------------------------------------------------------------
__global__ __launch_bounds__(256) void qkv_proj_kernel(
    const float* __restrict__ x,
    const float* __restrict__ Wq,
    const float* __restrict__ Wk,
    const float* __restrict__ Wv,
    unsigned short* __restrict__ q_ws,
    unsigned short* __restrict__ k_ws,
    unsigned short* __restrict__ v_ws)
{
  const int which = blockIdx.z;
  const float* W = (which == 0) ? Wq : (which == 1) ? Wk : Wv;
  unsigned short* outw = (which == 0) ? q_ws : (which == 1) ? k_ws : v_ws;

  const int m0 = blockIdx.y * 128;
  const int n0 = blockIdx.x * 64;
  const int tid  = threadIdx.x;
  const int lane = tid & 31;
  const int w    = tid >> 5;   // 0..7
  const int wm   = w >> 1;     // 0..3
  const int wn   = w & 1;      // 0..1
  const int ln   = lane & 15;
  const int hi   = lane >> 4;

  __shared__ unsigned int   As[128][16];   // 128 x 32 bf16 (k-pairs packed)
  __shared__ unsigned short Bs[64][32];    // transposed: [n][k]

  const v8f vzero = {0.f,0.f,0.f,0.f,0.f,0.f,0.f,0.f};
  v8f acc[2][2];
  for (int i = 0; i < 2; ++i) for (int j = 0; j < 2; ++j) acc[i][j] = vzero;

  for (int k0 = 0; k0 < MHA_D; k0 += 32) {
    {   // stage A tile (f32 -> bf16)
      int row = tid >> 1;
      int cb  = (tid & 1) * 16;
      const float* src = x + (size_t)(m0 + row) * MHA_D + k0 + cb;
      #pragma unroll
      for (int j = 0; j < 8; ++j)
        As[row][(cb >> 1) + j] = pack_bf16(src[2*j], src[2*j + 1]);
    }
    {   // stage B tile transposed (f32 -> bf16)
      int kk = tid >> 3;
      int nb = (tid & 7) * 8;
      const float* src = W + (size_t)(k0 + kk) * MHA_D + n0 + nb;
      #pragma unroll
      for (int j = 0; j < 8; ++j) Bs[nb + j][kk] = bf16_of(src[j]);
    }
    __syncthreads();

    Frag16 af[2], bfz[2];
    #pragma unroll
    for (int i = 0; i < 2; ++i) {
      int row = wm * 32 + i * 16 + ln;
      #pragma unroll
      for (int t = 0; t < 8; ++t)
        af[i].u[t] = As[row][(t < 4 ? t : t + 4) + hi * 4];
    }
    #pragma unroll
    for (int j = 0; j < 2; ++j) {
      int col = wn * 32 + j * 16 + ln;
      const unsigned int* bp = (const unsigned int*)&Bs[col][0];
      #pragma unroll
      for (int t = 0; t < 8; ++t) bfz[j].u[t] = bp[(t < 4 ? t : t + 4) + hi * 4];
    }
    #pragma unroll
    for (int i = 0; i < 2; ++i)
      #pragma unroll
      for (int j = 0; j < 2; ++j)
        acc[i][j] = __builtin_amdgcn_wmma_f32_16x16x32_bf16(
            false, af[i].v, false, bfz[j].v, (short)0, acc[i][j], false, false);
    __syncthreads();
  }

  #pragma unroll
  for (int i = 0; i < 2; ++i) {
    #pragma unroll
    for (int j = 0; j < 2; ++j) {
      int ncol = n0 + wn * 32 + j * 16 + ln;
      int h  = ncol >> 6;
      int hd = ncol & 63;
      #pragma unroll
      for (int r = 0; r < 8; ++r) {
        int mrow = m0 + wm * 32 + i * 16 + r + hi * 8;
        int bb = mrow >> 11, s = mrow & 2047;
        size_t idx = (((size_t)(bb * MHA_H + h)) * MHA_S + s) * MHA_HD + hd;
        outw[idx] = bf16_of(acc[i][j][r]);
      }
    }
  }
}

// ---------------------------------------------------------------------------
// Kernel 2: flash attention.  One block = (b, h, 64 q-rows).  4 waves, each
// owning 16 q-rows end-to-end.  K tile staged via async global->LDS.
// ---------------------------------------------------------------------------
__global__ __launch_bounds__(128) void attn_kernel(
    const unsigned short* __restrict__ q_ws,
    const unsigned short* __restrict__ k_ws,
    const unsigned short* __restrict__ v_ws,
    const int* __restrict__ att_mask,        // (B,S,S)
    unsigned short* __restrict__ ctx_ws)     // bf16 (B,S,D), col = h*64+hd
{
  const int q0 = blockIdx.x * 64;
  const int h  = blockIdx.y;
  const int b  = blockIdx.z;
  const int tid  = threadIdx.x;
  const int lane = tid & 31;
  const int w    = tid >> 5;   // 0..3
  const int ln   = lane & 15;
  const int hi   = lane >> 4;

  const size_t bh = (size_t)(b * MHA_H + h);
  const unsigned int* Qg = (const unsigned int*)(q_ws + bh * MHA_S * MHA_HD);
  const unsigned int* Kg = (const unsigned int*)(k_ws + bh * MHA_S * MHA_HD);
  const unsigned int* Vg = (const unsigned int*)(v_ws + bh * MHA_S * MHA_HD);
  const unsigned short* Vh = v_ws + bh * MHA_S * MHA_HD;
  const int* Mg = att_mask + (size_t)b * MHA_S * MHA_S;

  __shared__ unsigned int   Ks[64][32];     // K rows, d-pairs packed
  __shared__ unsigned short Vt[64][64];     // transposed: [d][kpos]
  __shared__ unsigned short Ps[4][16][64];  // per-wave P slab

  // preload Q fragments for this wave's 16 rows (2 d-steps of 32)
  Frag16 qa[2];
  {
    const unsigned int* qp = Qg + (size_t)(q0 + w * 16 + ln) * 32;
    #pragma unroll
    for (int ds = 0; ds < 2; ++ds)
      #pragma unroll
      for (int t = 0; t < 8; ++t)
        qa[ds].u[t] = qp[ds * 16 + (t < 4 ? t : t + 4) + hi * 4];
  }

  const v8f vzero = {0.f,0.f,0.f,0.f,0.f,0.f,0.f,0.f};
  v8f ctx[4];
  #pragma unroll
  for (int t = 0; t < 4; ++t) ctx[t] = vzero;
  float m_run[8], l_run[8];
  #pragma unroll
  for (int r = 0; r < 8; ++r) { m_run[r] = -1e30f; l_run[r] = 0.f; }

  const float scale = 0.125f;   // 1/sqrt(64)

  for (int k0 = 0; k0 < MHA_S; k0 += 64) {
    __syncthreads();   // previous iteration's LDS readers done

    // prefetch next k-step's mask rows into L2/L0 (global_prefetch_b8)
    if (k0 + 64 < MHA_S)
      __builtin_prefetch(Mg + (size_t)(q0 + (tid >> 1)) * MHA_S + k0 + 64, 0, 1);

    {   // stage K tile: 64 x 32 dwords (pure copy)
      int row = tid >> 1;
      int cb  = (tid & 1) * 16;
#if USE_ASYNC_LDS
      AS1 i32x4* src = (AS1 i32x4*)(Kg + (size_t)(k0 + row) * 32 + cb);
      AS3 i32x4* dst = (AS3 i32x4*)&Ks[row][cb];
      __builtin_amdgcn_global_load_async_to_lds_b128(src + 0, dst + 0, 0, 0);
      __builtin_amdgcn_global_load_async_to_lds_b128(src + 1, dst + 1, 0, 0);
      __builtin_amdgcn_global_load_async_to_lds_b128(src + 2, dst + 2, 0, 0);
      __builtin_amdgcn_global_load_async_to_lds_b128(src + 3, dst + 3, 0, 0);
#else
      const unsigned int* src = Kg + (size_t)(k0 + row) * 32 + cb;
      #pragma unroll
      for (int j = 0; j < 16; ++j) Ks[row][cb + j] = src[j];
#endif
    }
    {   // stage V transposed (needs element rearrangement -> VGPR path)
      int kp = tid >> 1;
      int db = (tid & 1) * 32;
      const unsigned int* src = Vg + (size_t)(k0 + kp) * 32 + (db >> 1);
      #pragma unroll
      for (int j = 0; j < 16; ++j) {
        unsigned int u = src[j];
        Vt[db + 2*j][kp]     = (unsigned short)(u & 0xFFFFu);
        Vt[db + 2*j + 1][kp] = (unsigned short)(u >> 16);
      }
    }
#if USE_ASYNC_LDS
    __builtin_amdgcn_s_wait_asynccnt(0);
#endif
    __syncthreads();

    // ---- scores = Q @ K^T (16 x 64), 2 d-steps x 4 k-tiles
    v8f sc[4];
    #pragma unroll
    for (int t = 0; t < 4; ++t) sc[t] = vzero;
    #pragma unroll
    for (int ds = 0; ds < 2; ++ds) {
      #pragma unroll
      for (int t = 0; t < 4; ++t) {
        Frag16 kb;
        #pragma unroll
        for (int j = 0; j < 8; ++j)
          kb.u[j] = Ks[t * 16 + ln][ds * 16 + (j < 4 ? j : j + 4) + hi * 4];
        sc[t] = __builtin_amdgcn_wmma_f32_16x16x32_bf16(
            false, qa[ds].v, false, kb.v, (short)0, sc[t], false, false);
      }
    }

    // ---- scale + mask
    #pragma unroll
    for (int t = 0; t < 4; ++t) {
      #pragma unroll
      for (int r = 0; r < 8; ++r) {
        int qrow = q0 + w * 16 + r + hi * 8;
        int kcol = k0 + t * 16 + ln;
        int mk = Mg[(size_t)qrow * MHA_S + kcol];
        float s = sc[t][r] * scale;
        sc[t][r] = mk ? s : -1e30f;
      }
    }

    // ---- online softmax (row lives in 16 lanes of one VGPR)
    #pragma unroll
    for (int r = 0; r < 8; ++r) {
      float mx = fmaxf(fmaxf(sc[0][r], sc[1][r]), fmaxf(sc[2][r], sc[3][r]));
      #pragma unroll
      for (int sh = 1; sh < 16; sh <<= 1)
        mx = fmaxf(mx, __shfl_xor(mx, sh, 32));
      float mnew = fmaxf(m_run[r], mx);
      float alpha = __expf(m_run[r] - mnew);
      m_run[r] = mnew;
      float rs = 0.f;
      #pragma unroll
      for (int t = 0; t < 4; ++t) {
        float p = __expf(sc[t][r] - mnew);
        sc[t][r] = p;
        rs += p;
      }
      #pragma unroll
      for (int sh = 1; sh < 16; sh <<= 1)
        rs += __shfl_xor(rs, sh, 32);
      l_run[r] = l_run[r] * alpha + rs;
      #pragma unroll
      for (int t = 0; t < 4; ++t) ctx[t][r] *= alpha;
    }

    // ---- P through per-wave LDS slab (C-layout -> A-layout)
    #pragma unroll
    for (int t = 0; t < 4; ++t)
      #pragma unroll
      for (int r = 0; r < 8; ++r)
        Ps[w][r + hi * 8][t * 16 + ln] = bf16_of(sc[t][r]);

    // ---- ctx += P @ V  (K-dim 64 -> 2 steps of 32)
    const unsigned int* prow = (const unsigned int*)&Ps[w][ln][0];
    #pragma unroll
    for (int kk = 0; kk < 2; ++kk) {
      Frag16 pa;
      #pragma unroll
      for (int t = 0; t < 8; ++t)
        pa.u[t] = prow[kk * 16 + (t < 4 ? t : t + 4) + hi * 4];
      #pragma unroll
      for (int t = 0; t < 4; ++t) {
        Frag16 vb;
        const unsigned int* vrow = (const unsigned int*)&Vt[t * 16 + ln][0];
        #pragma unroll
        for (int j = 0; j < 8; ++j) vb.u[j] = vrow[kk * 16 + (j < 4 ? j : j + 4) + hi * 4];
        ctx[t] = __builtin_amdgcn_wmma_f32_16x16x32_bf16(
            false, pa.v, false, vb.v, (short)0, ctx[t], false, false);
      }
    }
  }

  // ---- finalize: normalize; fully-masked rows get V[0,:] (reference rule)
  #pragma unroll
  for (int r = 0; r < 8; ++r) {
    bool dead = (m_run[r] <= -1e29f);
    float inv = dead ? 0.f : 1.f / l_run[r];
    int qrow = q0 + w * 16 + r + hi * 8;
    #pragma unroll
    for (int t = 0; t < 4; ++t) {
      int d = t * 16 + ln;
      float v = ctx[t][r] * inv;
      if (dead) v = __uint_as_float(((unsigned int)Vh[d]) << 16);
      size_t idx = ((size_t)b * MHA_S + qrow) * MHA_D + h * MHA_HD + d;
      ctx_ws[idx] = bf16_of(v);
    }
  }
}

// ---------------------------------------------------------------------------
// Kernel 3: output projection  out = ctx @ Wo + bo  (f32 out).
// A tile is a conversion-free 2D copy -> Tensor Data Mover (one D# per
// k-step, issued by wave 0, tracked with TENSORcnt).
// ---------------------------------------------------------------------------
__global__ __launch_bounds__(256) void out_proj_kernel(
    const unsigned short* __restrict__ ctx_ws,   // bf16 (4096,1024)
    const float* __restrict__ Wo,
    const float* __restrict__ bo,
    float* __restrict__ out)
{
  const int m0 = blockIdx.y * 128;
  const int n0 = blockIdx.x * 64;
  const int tid  = threadIdx.x;
  const int lane = tid & 31;
  const int w    = tid >> 5;
  const int wm   = w >> 1;
  const int wn   = w & 1;
  const int ln   = lane & 15;
  const int hi   = lane >> 4;

  __shared__ unsigned int   As[128][16];
  __shared__ unsigned short Bs[64][32];

  const v8f vzero = {0.f,0.f,0.f,0.f,0.f,0.f,0.f,0.f};
  v8f acc[2][2];
  for (int i = 0; i < 2; ++i) for (int j = 0; j < 2; ++j) acc[i][j] = vzero;

  const unsigned int* ctxd = (const unsigned int*)ctx_ws;

  for (int k0 = 0; k0 < MHA_D; k0 += 32) {
#if USE_TDM
    if (w == 0) {   // one TDM descriptor moves the whole 128x64B A tile
      unsigned lds_off = (unsigned)(unsigned long long)(AS3 unsigned int*)&As[0][0];
      unsigned long long ga =
          (unsigned long long)(ctxd + (size_t)m0 * (MHA_D / 2) + (k0 >> 1));
      u32x4 g0;
      g0[0] = 1u;                                        // count=1 (user D#)
      g0[1] = lds_off;                                   // lds_addr (bytes)
      g0[2] = (unsigned)ga;                              // global_addr lo
      g0[3] = (unsigned)((ga >> 32) & 0x1FFFFFFu) | (2u << 30); // hi + type=2
      i32x8 g1;
      g1[0] = (int)(2u << 16);                           // data_size=4B
      g1[1] = (int)(((MHA_D / 2) & 0xFFFFu) << 16);      // tensor_dim0 lo16
      g1[2] = (int)(((MHA_D / 2) >> 16) | ((4096u & 0xFFFFu) << 16)); // dim0 hi / dim1 lo
      g1[3] = (int)((4096u >> 16) | (16u << 16));        // dim1 hi / tile_dim0=16
      g1[4] = (int)128;                                  // tile_dim1=128, tile_dim2=0
      g1[5] = (int)(MHA_D / 2);                          // tensor_dim0_stride lo32
      g1[6] = 0;
      g1[7] = 0;
      i32x4 gz = {0, 0, 0, 0};
#if __clang_major__ >= 23
      i32x8 gz8 = {0, 0, 0, 0, 0, 0, 0, 0};
      __builtin_amdgcn_tensor_load_to_lds(g0, g1, gz, gz, gz8, 0);
#else
      __builtin_amdgcn_tensor_load_to_lds(g0, g1, gz, gz, 0);
#endif
      __builtin_amdgcn_s_wait_tensorcnt(0);
    }
#elif USE_ASYNC_LDS
    {
      int row = tid >> 1;
      int cb  = (tid & 1) * 8;
      AS1 i32x4* src = (AS1 i32x4*)(ctxd +
          (size_t)(m0 + row) * (MHA_D / 2) + (k0 >> 1) + cb);
      AS3 i32x4* dst = (AS3 i32x4*)&As[row][cb];
      __builtin_amdgcn_global_load_async_to_lds_b128(src + 0, dst + 0, 0, 0);
      __builtin_amdgcn_global_load_async_to_lds_b128(src + 1, dst + 1, 0, 0);
    }
#else
    {   // fallback: VGPR copy
      int row = tid >> 1;
      int cb  = (tid & 1) * 8;
      const unsigned int* src = ctxd + (size_t)(m0 + row) * (MHA_D / 2) + (k0 >> 1) + cb;
      #pragma unroll
      for (int j = 0; j < 8; ++j) As[row][cb + j] = src[j];
    }
#endif
    {   // stage B transposed (f32 -> bf16)
      int kk = tid >> 3;
      int nb = (tid & 7) * 8;
      const float* src = Wo + (size_t)(k0 + kk) * MHA_D + n0 + nb;
      #pragma unroll
      for (int j = 0; j < 8; ++j) Bs[nb + j][kk] = bf16_of(src[j]);
    }
#if !USE_TDM && USE_ASYNC_LDS
    __builtin_amdgcn_s_wait_asynccnt(0);
#endif
    __syncthreads();

    Frag16 af[2], bfz[2];
    #pragma unroll
    for (int i = 0; i < 2; ++i) {
      int row = wm * 32 + i * 16 + ln;
      #pragma unroll
      for (int t = 0; t < 8; ++t)
        af[i].u[t] = As[row][(t < 4 ? t : t + 4) + hi * 4];
    }
    #pragma unroll
    for (int j = 0; j < 2; ++j) {
      int col = wn * 32 + j * 16 + ln;
      const unsigned int* bp = (const unsigned int*)&Bs[col][0];
      #pragma unroll
      for (int t = 0; t < 8; ++t) bfz[j].u[t] = bp[(t < 4 ? t : t + 4) + hi * 4];
    }
    #pragma unroll
    for (int i = 0; i < 2; ++i)
      #pragma unroll
      for (int j = 0; j < 2; ++j)
        acc[i][j] = __builtin_amdgcn_wmma_f32_16x16x32_bf16(
            false, af[i].v, false, bfz[j].v, (short)0, acc[i][j], false, false);
    __syncthreads();
  }

  #pragma unroll
  for (int i = 0; i < 2; ++i) {
    #pragma unroll
    for (int j = 0; j < 2; ++j) {
      int ncol = n0 + wn * 32 + j * 16 + ln;
      float bias = bo[ncol];
      #pragma unroll
      for (int r = 0; r < 8; ++r) {
        int mrow = m0 + wm * 32 + i * 16 + r + hi * 8;
        out[(size_t)mrow * MHA_D + ncol] = acc[i][j][r] + bias;
      }
    }
  }
}

// ---------------------------------------------------------------------------
extern "C" void kernel_launch(void* const* d_in, const int* in_sizes, int n_in,
                              void* d_out, int out_size, void* d_ws, size_t ws_size,
                              hipStream_t stream) {
  const float* x    = (const float*)d_in[0];
  const int*   mask = (const int*)d_in[1];
  const float* Wq   = (const float*)d_in[2];
  const float* Wk   = (const float*)d_in[3];
  const float* Wv   = (const float*)d_in[4];
  const float* Wo   = (const float*)d_in[5];
  const float* bo   = (const float*)d_in[6];
  float* out = (float*)d_out;

  const size_t qkv_elems = (size_t)2 * MHA_H * MHA_S * MHA_HD;  // 4M bf16 each
  unsigned short* q_ws = (unsigned short*)d_ws;
  unsigned short* k_ws = q_ws + qkv_elems;
  unsigned short* v_ws = k_ws + qkv_elems;
  unsigned short* c_ws = v_ws + qkv_elems;

  dim3 g1(MHA_D / 64, (2 * MHA_S) / 128, 3);
  qkv_proj_kernel<<<g1, dim3(256), 0, stream>>>(x, Wq, Wk, Wv, q_ws, k_ws, v_ws);

  dim3 g2(MHA_S / 64, MHA_H, 2);
  attn_kernel<<<g2, dim3(128), 0, stream>>>(q_ws, k_ws, v_ws, mask, c_ws);

  dim3 g3(MHA_D / 64, (2 * MHA_S) / 128, 1);
  out_proj_kernel<<<g3, dim3(256), 0, stream>>>(c_ws, Wo, bo, out);
}
